// RNN_71725953843375
// MI455X (gfx1250) — matrix operand order repaired and checked by reading, not compiled, not run on hardware
//
#include <hip/hip_runtime.h>
#include <hip/hip_bf16.h>

typedef __attribute__((ext_vector_type(16))) __bf16 v16bf;
typedef __attribute__((ext_vector_type(8)))  float  v8f;
typedef __attribute__((ext_vector_type(4)))  unsigned int u32x4;

#define BSZ  128
#define TLEN 512
#define EDIM 256
#define HDIM 512
#define OUTD 9
#define NBLK 32   // one block per N-tile: 32 blocks * 8 waves(M-tiles) = 256 tiles

union FragA {
  v16bf v;
  __bf16 h[16];
  u32x4 q[2];
};

#define WMMA_BF16(A, Bf, C) \
  __builtin_amdgcn_wmma_f32_16x16x32_bf16(false, (A), false, (Bf), (short)0, (C), false, false)

// fast tanh via the transcendental unit (co-executes with WMMA on CDNA5)
__device__ __forceinline__ float fast_tanh(float x) {
  float ax = fabsf(x);
  float e  = __expf(ax + ax);          // v_exp_f32
  float r  = 1.0f - 2.0f / (e + 1.0f); // v_rcp_f32 + fma
  return copysignf(r, x);
}

// ---------------------------------------------------------------------------
// Pack a K-stacked B panel [Ktot x 512] (B[k][n] = src[n][k]) into per-lane
// WMMA B-fragment layout: dst[((nt*KT+kt)*32+lane)*16 + j] = bf16(B[kbase+(j<8?j:8+j)][n])
// where kbase = kt*32 + (lane>=16 ? 8 : 0), n = nt*16 + (lane&15).
// ---------------------------------------------------------------------------
__global__ void pack_b(const float* __restrict__ s0, int ld0, int K0,
                       const float* __restrict__ s1, int ld1,
                       int KT, int NT, int Nvalid,
                       __bf16* __restrict__ dst) {
  int id = blockIdx.x * blockDim.x + threadIdx.x;
  int total = KT * NT * 32;
  if (id >= total) return;
  int lane = id & 31;
  int tile = id >> 5;            // tile = nt*KT + kt
  int kt = tile % KT;
  int nt = tile / KT;
  int n  = nt * 16 + (lane & 15);
  int kb = kt * 32 + (lane >> 4) * 8;
  __bf16* d = dst + (size_t)tile * 512 + lane * 16;
  #pragma unroll
  for (int j = 0; j < 16; ++j) {
    int k = kb + (j < 8 ? j : 8 + j);
    float v = 0.f;
    if (n < Nvalid)
      v = (k < K0) ? s0[(size_t)n * ld0 + k] : s1[(size_t)n * ld1 + (k - K0)];
    d[j] = (__bf16)v;
  }
}

// zero recurrent state, build combined biases, reset the grid barrier counter
__global__ void init_misc(const float* __restrict__ Wb, const float* __restrict__ Ukb,
                          const float* __restrict__ Wkb, const float* __restrict__ Vb,
                          float* __restrict__ bias1, float* __restrict__ bias2,
                          float* __restrict__ biasY,
                          unsigned int* __restrict__ zzero, int zwords,
                          unsigned* __restrict__ cnt) {
  int id = blockIdx.x * blockDim.x + threadIdx.x;
  if (id < zwords) zzero[id] = 0u;
  if (id < HDIM) { bias1[id] = Wb[id] + Ukb[id]; bias2[id] = Wkb[id] + Ukb[id]; }
  if (id < 16)   biasY[id] = (id < OUTD) ? Vb[id] : 0.f;
  if (id == 0)   *cnt = 0u;
}

// device-scope grid barrier (monotone counter, no reset needed within a launch)
__device__ __forceinline__ void gbar(unsigned* cnt, unsigned target) {
  __threadfence();
  __syncthreads();
  if (threadIdx.x == 0) {
    __hip_atomic_fetch_add(cnt, 1u, __ATOMIC_RELAXED, __HIP_MEMORY_SCOPE_AGENT);
    while (__hip_atomic_load(cnt, __ATOMIC_RELAXED, __HIP_MEMORY_SCOPE_AGENT) < target)
      __builtin_amdgcn_s_sleep(2);
    __threadfence();
  }
  __syncthreads();
}

// ---------------------------------------------------------------------------
// Persistent scan kernel. Block = one N-tile; wave = one M-tile.
// Weight panels live in LDS for the whole scan (staged once).
// ---------------------------------------------------------------------------
__global__ __launch_bounds__(256, 1) void rnn_scan(
    const float* __restrict__ emb,
    const __bf16* __restrict__ B1p,   // [768 x 512]  = [W^T ; U0^T] packed
    const __bf16* __restrict__ B2p,   // [1024 x 512] = [Wk0^T ; U0^T] packed
    const __bf16* __restrict__ BVp,   // [512 x 16]   = V^T packed (9 valid cols)
    const float* __restrict__ bias1, const float* __restrict__ bias2,
    const float* __restrict__ biasY,
    __bf16* __restrict__ z1a, __bf16* __restrict__ z1b,
    __bf16* __restrict__ z2a, __bf16* __restrict__ z2b,
    unsigned* cnt, float* __restrict__ out) {
  // LDS-resident weight panels for this block's N-tile (16B-aligned for b128)
  __shared__ u32x4 sB1[24 * 64];   // 24 KB: layer-1 panel, 24 k-tiles
  __shared__ u32x4 sB2[32 * 64];   // 32 KB: layer-2 panel, 32 k-tiles
  __shared__ u32x4 sBV[16 * 64];   // 16 KB: output panel (used by block 0 only)

  const int lane = threadIdx.x & 31;
  const int mt   = threadIdx.x >> 5;          // 0..7   (M tile of 128 rows)
  const int nt   = blockIdx.x;                // 0..31  (N tile of 512 cols)
  const int m0   = mt * 16, n0 = nt * 16;
  const int hh   = lane >> 4;                 // lane half
  const int lm   = lane & 15;
  const int arow = m0 + lm;                   // A-fragment source row for this lane

  // ---- stage weight panels into LDS once ----
  {
    const u32x4* s1 = (const u32x4*)(B1p + (size_t)nt * 24 * 512);
    for (int i = threadIdx.x; i < 24 * 64; i += 256) sB1[i] = s1[i];
    const u32x4* s2 = (const u32x4*)(B2p + (size_t)nt * 32 * 512);
    for (int i = threadIdx.x; i < 32 * 64; i += 256) sB2[i] = s2[i];
    if (nt == 0) {
      const u32x4* sv = (const u32x4*)BVp;
      for (int i = threadIdx.x; i < 16 * 64; i += 256) sBV[i] = sv[i];
    }
    __syncthreads();
  }

  const float bb1 = bias1[n0 + lm];
  const float bb2 = bias2[n0 + lm];
  const float bby = biasY[lm];
  const v8f vzero = {};

  unsigned target = 0;

  for (int t = 0; t < TLEN; ++t) {
    __bf16*       z1w = (t & 1) ? z1b : z1a;
    const __bf16* z1r = (t & 1) ? z1a : z1b;
    __bf16*       z2w = (t & 1) ? z2b : z2a;
    const __bf16* z2r = (t & 1) ? z2a : z2b;

    // ---- layer 1: z1_t = tanh([x_t | z1_{t-1}] @ B1 + b1), K = 256+512 ----
    v8f acc0 = vzero, acc1 = vzero;
    {
      const float*  xrow = emb + (size_t)arow * (TLEN * EDIM) + (size_t)t * EDIM;
      const __bf16* zrow = z1r + (size_t)arow * HDIM;
      if (t + 1 < TLEN) __builtin_prefetch(xrow + EDIM, 0, 0);  // next step's x panel
      #pragma unroll 2
      for (int kt = 0; kt < 8; ++kt) {                       // embedding part (hw cvt)
        FragA a, b;
        const float* p = xrow + kt * 32 + hh * 8;
        #pragma unroll
        for (int i = 0; i < 8; ++i) { a.h[i] = (__bf16)p[i]; a.h[8 + i] = (__bf16)p[16 + i]; }
        b.q[0] = sB1[(kt * 32 + lane) * 2];
        b.q[1] = sB1[(kt * 32 + lane) * 2 + 1];
        if (kt & 1) acc1 = WMMA_BF16(a.v, b.v, acc1); else acc0 = WMMA_BF16(a.v, b.v, acc0);
      }
      #pragma unroll 4
      for (int kt = 8; kt < 24; ++kt) {                      // recurrent part (bf16)
        FragA a, b;
        const __bf16* p = zrow + (kt - 8) * 32 + hh * 8;
        a.q[0] = *(const u32x4*)p; a.q[1] = *(const u32x4*)(p + 16);
        b.q[0] = sB1[(kt * 32 + lane) * 2];
        b.q[1] = sB1[(kt * 32 + lane) * 2 + 1];
        if (kt & 1) acc1 = WMMA_BF16(a.v, b.v, acc1); else acc0 = WMMA_BF16(a.v, b.v, acc0);
      }
    }
    {
      __bf16* zo = z1w + (size_t)(n0 + lm);
      #pragma unroll
      for (int r = 0; r < 8; ++r)
        zo[(size_t)(m0 + hh * 8 + r) * HDIM] = (__bf16)fast_tanh(acc0[r] + acc1[r] + bb1);
    }
    gbar(cnt, target += NBLK);

    // ---- layer 2: z2_t = tanh([z1_t | z2_{t-1}] @ B2 + b2), K = 512+512 ----
    acc0 = vzero; acc1 = vzero;
    {
      const __bf16* z1row = z1w + (size_t)arow * HDIM;
      const __bf16* z2row = z2r + (size_t)arow * HDIM;
      #pragma unroll 4
      for (int kt = 0; kt < 16; ++kt) {                      // z1_t part
        FragA a, b;
        const __bf16* p = z1row + kt * 32 + hh * 8;
        a.q[0] = *(const u32x4*)p; a.q[1] = *(const u32x4*)(p + 16);
        b.q[0] = sB2[(kt * 32 + lane) * 2];
        b.q[1] = sB2[(kt * 32 + lane) * 2 + 1];
        if (kt & 1) acc1 = WMMA_BF16(a.v, b.v, acc1); else acc0 = WMMA_BF16(a.v, b.v, acc0);
      }
      #pragma unroll 4
      for (int kt = 16; kt < 32; ++kt) {                     // z2_{t-1} part
        FragA a, b;
        const __bf16* p = z2row + (kt - 16) * 32 + hh * 8;
        a.q[0] = *(const u32x4*)p; a.q[1] = *(const u32x4*)(p + 16);
        b.q[0] = sB2[(kt * 32 + lane) * 2];
        b.q[1] = sB2[(kt * 32 + lane) * 2 + 1];
        if (kt & 1) acc1 = WMMA_BF16(a.v, b.v, acc1); else acc0 = WMMA_BF16(a.v, b.v, acc0);
      }
    }
    {
      __bf16* zo = z2w + (size_t)(n0 + lm);
      #pragma unroll
      for (int r = 0; r < 8; ++r)
        zo[(size_t)(m0 + hh * 8 + r) * HDIM] = (__bf16)fast_tanh(acc0[r] + acc1[r] + bb2);
    }
    gbar(cnt, target += NBLK);

    // ---- output: y_t = z2_t @ V^T + V_b (block 0 only; block-uniform branch) ----
    if (nt == 0) {
      v8f ya = vzero, yb = vzero;
      const __bf16* zrow = z2w + (size_t)arow * HDIM;
      #pragma unroll 4
      for (int kt = 0; kt < 16; ++kt) {
        FragA a, b;
        const __bf16* p = zrow + kt * 32 + hh * 8;
        a.q[0] = *(const u32x4*)p; a.q[1] = *(const u32x4*)(p + 16);
        b.q[0] = sBV[(kt * 32 + lane) * 2];
        b.q[1] = sBV[(kt * 32 + lane) * 2 + 1];
        if (kt & 1) yb = WMMA_BF16(a.v, b.v, yb); else ya = WMMA_BF16(a.v, b.v, ya);
      }
      if (lm < OUTD) {
        #pragma unroll
        for (int r = 0; r < 8; ++r) {
          int bidx = m0 + hh * 8 + r;
          out[(size_t)bidx * (TLEN * OUTD) + (size_t)t * OUTD + lm] = ya[r] + yb[r] + bby;
        }
      }
    }
  }
}

extern "C" void kernel_launch(void* const* d_in, const int* in_sizes, int n_in,
                              void* d_out, int out_size, void* d_ws, size_t ws_size,
                              hipStream_t stream) {
  (void)in_sizes; (void)n_in; (void)out_size; (void)ws_size;
  const float* emb  = (const float*)d_in[0];   // [128,512,256]
  const float* W_w  = (const float*)d_in[1];   // [512,256]
  const float* W_b  = (const float*)d_in[2];   // [512]
  const float* Uk_w = (const float*)d_in[3];   // [2,512,512] (only slice 0 used)
  const float* Uk_b = (const float*)d_in[4];   // [2,512]     (only row 0 used)
  const float* Wk_w = (const float*)d_in[5];   // [1,512,512]
  const float* Wk_b = (const float*)d_in[6];   // [1,512]
  const float* V_w  = (const float*)d_in[7];   // [9,512]
  const float* V_b  = (const float*)d_in[8];   // [9]
  float* out = (float*)d_out;                  // [128,512,9]

  char* w = (char*)d_ws;                       // ~2.3 MB total
  unsigned* cnt   = (unsigned*)w;
  float*    bias1 = (float*)(w + 1024);
  float*    bias2 = (float*)(w + 1024 + 4096);
  float*    biasY = (float*)(w + 1024 + 8192);
  __bf16*   B1p   = (__bf16*)(w + 16384);                       // 768 KB
  __bf16*   B2p   = (__bf16*)(w + 16384 + 786432);              // 1 MB
  __bf16*   BVp   = (__bf16*)(w + 16384 + 786432 + 1048576);    // 16 KB
  __bf16*   z1a   = (__bf16*)(w + 16384 + 786432 + 1048576 + 16384);
  __bf16*   z1b   = z1a + 65536;
  __bf16*   z2a   = z1b + 65536;
  __bf16*   z2b   = z2a + 65536;

  // init: zero z state (4 x 128KB), combined biases, barrier counter
  init_misc<<<(131072 + 255) / 256, 256, 0, stream>>>(
      W_b, Uk_b, Wk_b, V_b, bias1, bias2, biasY, (unsigned int*)z1a, 131072, cnt);

  // pack B panels into per-lane fragment layout (bf16)
  pack_b<<<(24 * 32 * 32 + 255) / 256, 256, 0, stream>>>(
      W_w, EDIM, EDIM, Uk_w, HDIM, 24, 32, HDIM, B1p);          // [W^T ; U0^T]
  pack_b<<<(32 * 32 * 32 + 255) / 256, 256, 0, stream>>>(
      Wk_w, HDIM, HDIM, Uk_w, HDIM, 32, 32, HDIM, B2p);         // [Wk0^T ; U0^T]
  pack_b<<<(16 * 1 * 32 + 255) / 256, 256, 0, stream>>>(
      V_w, HDIM, HDIM, V_w, HDIM, 16, 1, OUTD, BVp);            // V^T (pad to 16 cols)

  // persistent scan: 32 blocks x 8 waves, 2 grid barriers per timestep
  rnn_scan<<<NBLK, 256, 0, stream>>>(emb, B1p, B2p, BVp, bias1, bias2, biasY,
                                     z1a, z1b, z2a, z2b, cnt, out);
}
